// Encoder_8134668059285
// MI455X (gfx1250) — matrix operand order repaired and checked by reading, not compiled, not run on hardware
//
#include <hip/hip_runtime.h>
#include <hip/hip_bf16.h>
#include <stdint.h>

// ---------------------------------------------------------------------------
// 3-layer GRU (PyTorch gate order r,z,n) for MI455X (gfx1250, wave32, WMMA).
//
//  * bf16 WMMA 16x16x32, fp32 accumulation; fused x-proj + recurrence per step.
//  * LDS double-buffered staging via gfx1250 async DMA
//    (global_load_async_to_lds_b128 + s_wait_asynccnt); 64-K slabs =>
//    12 WMMAs per sync. ds_load_b128 fragment reads, 144B row stride
//    (conflict-free: 36-bank step, 16 distinct banks per lane group).
//  * Split-phase K loops + DIN template: fixed accumulator registers per loop.
//  * Batched fragment loads per 32-K sub-step: all ds_loads issued before the
//    six independent WMMAs -> deep DS pipelining instead of wait-per-WMMA.
//  * Weights bf16 (~33MB) resident in the 192MB L2 across all 270 steps.
// ---------------------------------------------------------------------------

#define T_STEPS 90
#define BATCH   512
#define IN_DIM  256
#define HID     1024
#define LAYERS  3

// LDS row stride: 64 K + pad -> 72 bf16 (144B), 16B aligned, bank-conflict-free.
#define BPAD 72

typedef __attribute__((ext_vector_type(16))) __bf16 v16bf;
typedef __attribute__((ext_vector_type(8)))  __bf16 v8bf;
typedef __attribute__((ext_vector_type(8)))  float  v8f;

static __device__ inline v16bf cat8(v8bf lo, v8bf hi) {
    return __builtin_shufflevector(lo, hi, 0,1,2,3,4,5,6,7,8,9,10,11,12,13,14,15);
}

static __device__ inline v8f bfwmma(v16bf a, v16bf b, v8f c) {
    return __builtin_amdgcn_wmma_f32_16x16x32_bf16(
        false, a, false, b, (short)0, c, false, false);
}

static __device__ inline float sigmoidf_(float x) {
    return 1.0f / (1.0f + __expf(-x));
}

// gfx1250 async DMA: copy 16B global -> LDS, tracked by ASYNCcnt.
static __device__ inline void async_ld16(unsigned ldsByteAddr, const void* g) {
    unsigned long long ga = (unsigned long long)(size_t)g;
    asm volatile("global_load_async_to_lds_b128 %0, %1, off"
                 :: "v"(ldsByteAddr), "v"(ga) : "memory");
}

static __device__ inline void wait_async0() {
    asm volatile("s_wait_asynccnt 0x0" ::: "memory");
}

// ---------------------------------------------------------------------------
__global__ void f32_to_bf16_kernel(const float* __restrict__ in,
                                   __bf16* __restrict__ out, long n) {
    long i = (long)blockIdx.x * blockDim.x + threadIdx.x;
    long stride = (long)gridDim.x * blockDim.x;
    for (; i < n; i += stride) out[i] = (__bf16)in[i];
}

__global__ void init_h_kernel(const float* __restrict__ h0,
                              float* __restrict__ hf,
                              __bf16* __restrict__ hb, int n) {
    int i = blockIdx.x * blockDim.x + threadIdx.x;
    if (i < n) { float v = h0[i]; hf[i] = v; hb[i] = (__bf16)v; }
}

// ---------------------------------------------------------------------------
// Stage one 64-K slab into LDS with async DMA:
//   B tile: 192 gate-cols x 64 K = 1536 x 16B chunks (6 / thread)
//   A tile:  64 rows     x 64 K =  512 x 16B chunks (2 / thread)
// ---------------------------------------------------------------------------
static __device__ inline void stage_tiles(int tid, __bf16* tA, __bf16* tB,
                                          const __bf16* __restrict__ Asrc, int lda, int mBase,
                                          const __bf16* __restrict__ W,    int ldb, int nBase,
                                          int k) {
#pragma unroll
    for (int s = 0; s < 6; ++s) {
        const int ch   = tid + s * 256;        // 0..1535
        const int c    = ch >> 3;              // gate-col 0..191
        const int p    = ch & 7;               // 16B part of 64-K row
        const int gate = c >> 6;
        const int col  = c & 63;
        const __bf16* g = W + ((size_t)(gate * HID + nBase + col)) * ldb + k + p * 8;
        async_ld16((unsigned)(size_t)&tB[c * BPAD + p * 8], g);
    }
#pragma unroll
    for (int s = 0; s < 2; ++s) {
        const int ch = tid + s * 256;          // 0..511
        const int ra = ch >> 3;                // row 0..63
        const int p  = ch & 7;
        const __bf16* g = Asrc + (size_t)(mBase + ra) * lda + k + p * 8;
        async_ld16((unsigned)(size_t)&tA[ra * BPAD + p * 8], g);
    }
}

// ---------------------------------------------------------------------------
// One staged 64-K slab -> 12 WMMAs. Per 32-K sub-step: gather A + all 6 B
// fragments first (14 ds_load_b128 in flight), then 6 independent WMMAs.
// ---------------------------------------------------------------------------
static __device__ inline void compute_slab(const __bf16* tA, const __bf16* tB,
                                           int waveRow, int waveCol, int laneHalf, int lm,
                                           v8f (&ar)[2], v8f (&az)[2], v8f (&an)[2]) {
#pragma unroll
    for (int sub = 0; sub < 2; ++sub) {
        const int ks = sub * 32;

        // ---- gather all fragments for this sub-step ----
        const __bf16* aRow = tA + (waveRow * 16 + lm) * BPAD + ks;
        v16bf a = cat8(*(const v8bf*)(aRow + 16 * laneHalf),
                       *(const v8bf*)(aRow + 16 * laneHalf + 8));

        v16bf Bf[2][3];
#pragma unroll
        for (int nb = 0; nb < 2; ++nb) {
            const int c = waveCol * 32 + nb * 16 + lm;
#pragma unroll
            for (int g = 0; g < 3; ++g) {
                const __bf16* p = tB + (g * 64 + c) * BPAD + ks + 16 * laneHalf;
                Bf[nb][g] = cat8(*(const v8bf*)p, *(const v8bf*)(p + 8));
            }
        }

        // ---- six independent WMMAs (disjoint accumulators, no hazards) ----
#pragma unroll
        for (int nb = 0; nb < 2; ++nb) {
            ar[nb] = bfwmma(a, Bf[nb][0], ar[nb]);
            az[nb] = bfwmma(a, Bf[nb][1], az[nb]);
            an[nb] = bfwmma(a, Bf[nb][2], an[nb]);
        }
    }
}

// ---------------------------------------------------------------------------
// Fused GRU step. Grid (BATCH/64, HID/64), 256 threads = 8 waves (4 row x 2 col),
// wave tile 16 rows x 32 cols. DIN templated -> compile-time pipeline.
// ---------------------------------------------------------------------------
template <int DIN>
__global__ __launch_bounds__(256)
void gru_step_kernel(const __bf16* __restrict__ xin,   // [BATCH, DIN] (t-offset)
                     const __bf16* __restrict__ wih,   // [3H, DIN] bf16
                     const __bf16* __restrict__ whh,   // [3H, H]   bf16
                     const float*  __restrict__ bih,
                     const float*  __restrict__ bhh,
                     const float*  __restrict__ hin_f, // [BATCH, H] fp32
                     const __bf16* __restrict__ hin_b, // [BATCH, H] bf16
                     float*  __restrict__ hout_f,
                     __bf16* __restrict__ hout_b,
                     __bf16* seq_bf, float* seq_f, float* hid_out)
{
    __shared__ __bf16 tileA[2][64  * BPAD];   // 18 KB
    __shared__ __bf16 tileB[2][192 * BPAD];   // 54 KB

    const int tid      = threadIdx.x;
    const int lane     = tid & 31;
    const int wave     = tid >> 5;
    const int waveRow  = wave >> 1;
    const int waveCol  = wave & 1;
    const int mBase    = blockIdx.x * 64;
    const int nBase    = blockIdx.y * 64;
    const int m0       = mBase + waveRow * 16;
    const int laneHalf = lane >> 4;
    const int lm       = lane & 15;

    const int nc[2] = { waveCol * 32 + lm, waveCol * 32 + 16 + lm };

    v8f acc_r[2]  = { v8f{}, v8f{} };
    v8f acc_z[2]  = { v8f{}, v8f{} };
    v8f acc_xn[2] = { v8f{}, v8f{} };
    v8f acc_hn[2] = { v8f{}, v8f{} };

    constexpr int NX = DIN / 64;
    constexpr int NH = HID / 64;

    // Prologue: stage x slab 0.
    stage_tiles(tid, tileA[0], tileB[0], xin, DIN, mBase, wih, DIN, nBase, 0);

    int buf = 0;

    // ---- phase 1: x @ Wih^T ----
#pragma unroll 2
    for (int i = 0; i < NX; ++i) {
        wait_async0();
        __syncthreads();
        if (i + 1 < NX)
            stage_tiles(tid, tileA[buf ^ 1], tileB[buf ^ 1],
                        xin, DIN, mBase, wih, DIN, nBase, (i + 1) * 64);
        else
            stage_tiles(tid, tileA[buf ^ 1], tileB[buf ^ 1],
                        hin_b, HID, mBase, whh, HID, nBase, 0);
        compute_slab(tileA[buf], tileB[buf], waveRow, waveCol, laneHalf, lm,
                     acc_r, acc_z, acc_xn);
        buf ^= 1;
    }

    // ---- phase 2: h @ Whh^T ----
#pragma unroll 2
    for (int i = 0; i < NH; ++i) {
        wait_async0();
        __syncthreads();
        if (i + 1 < NH)
            stage_tiles(tid, tileA[buf ^ 1], tileB[buf ^ 1],
                        hin_b, HID, mBase, whh, HID, nBase, (i + 1) * 64);
        compute_slab(tileA[buf], tileB[buf], waveRow, waveCol, laneHalf, lm,
                     acc_r, acc_z, acc_hn);
        buf ^= 1;
    }

    // ---- epilogue: gates + state update ----
#pragma unroll
    for (int nb = 0; nb < 2; ++nb) {
        const int n = nBase + nc[nb];
        const float br_ = bih[n]       + bhh[n];
        const float bz_ = bih[HID + n] + bhh[HID + n];
        const float bin = bih[2 * HID + n];
        const float bhn = bhh[2 * HID + n];
#pragma unroll
        for (int r = 0; r < 8; ++r) {
            const int m = m0 + r + 8 * laneHalf;
            const float rv = sigmoidf_(acc_r[nb][r] + br_);
            const float zv = sigmoidf_(acc_z[nb][r] + bz_);
            const float nv = tanhf(acc_xn[nb][r] + bin + rv * (acc_hn[nb][r] + bhn));
            const float hp = hin_f[(size_t)m * HID + n];
            const float hn = (1.0f - zv) * nv + zv * hp;
            const size_t oh = (size_t)m * HID + n;
            hout_f[oh] = hn;
            hout_b[oh] = (__bf16)hn;
            if (seq_bf)  seq_bf[oh] = (__bf16)hn;
            if (seq_f)   seq_f[oh]  = hn;
            if (hid_out) hid_out[oh] = hn;
        }
    }
}

// ---------------------------------------------------------------------------
// Host launcher
// ---------------------------------------------------------------------------
extern "C" void kernel_launch(void* const* d_in, const int* in_sizes, int n_in,
                              void* d_out, int out_size, void* d_ws, size_t ws_size,
                              hipStream_t stream) {
    (void)in_sizes; (void)n_in; (void)out_size; (void)ws_size;

    const float* x  = (const float*)d_in[0];
    const float* h0 = (const float*)d_in[1];
    const float* wih_f[LAYERS] = { (const float*)d_in[2], (const float*)d_in[6],  (const float*)d_in[10] };
    const float* whh_f[LAYERS] = { (const float*)d_in[3], (const float*)d_in[7],  (const float*)d_in[11] };
    const float* bih[LAYERS]   = { (const float*)d_in[4], (const float*)d_in[8],  (const float*)d_in[12] };
    const float* bhh[LAYERS]   = { (const float*)d_in[5], (const float*)d_in[9],  (const float*)d_in[13] };

    float* out_seq = (float*)d_out;
    float* out_hid = out_seq + (size_t)T_STEPS * BATCH * HID;

    char* base = (char*)d_ws;
    size_t off = 0;
    auto carve = [&](size_t bytes) -> char* {
        off = (off + 255) & ~(size_t)255;
        char* p = base + off;
        off += bytes;
        return p;
    };

    __bf16* x_bf = (__bf16*)carve((size_t)T_STEPS * BATCH * IN_DIM * 2);
    __bf16* wih_b[LAYERS];
    __bf16* whh_b[LAYERS];
    const size_t wih_elems[LAYERS] = { (size_t)3 * HID * IN_DIM, (size_t)3 * HID * HID, (size_t)3 * HID * HID };
    for (int l = 0; l < LAYERS; ++l) {
        wih_b[l] = (__bf16*)carve(wih_elems[l] * 2);
        whh_b[l] = (__bf16*)carve((size_t)3 * HID * HID * 2);
    }
    __bf16* seqA = (__bf16*)carve((size_t)T_STEPS * BATCH * HID * 2);
    __bf16* seqB = (__bf16*)carve((size_t)T_STEPS * BATCH * HID * 2);
    float*  h_f[2];
    __bf16* h_b[2];
    h_f[0] = (float*)carve((size_t)BATCH * HID * 4);
    h_f[1] = (float*)carve((size_t)BATCH * HID * 4);
    h_b[0] = (__bf16*)carve((size_t)BATCH * HID * 2);
    h_b[1] = (__bf16*)carve((size_t)BATCH * HID * 2);

    {
        long n = (long)T_STEPS * BATCH * IN_DIM;
        f32_to_bf16_kernel<<<2048, 256, 0, stream>>>(x, x_bf, n);
        for (int l = 0; l < LAYERS; ++l) {
            f32_to_bf16_kernel<<<2048, 256, 0, stream>>>(wih_f[l], wih_b[l], (long)wih_elems[l]);
            f32_to_bf16_kernel<<<2048, 256, 0, stream>>>(whh_f[l], whh_b[l], (long)3 * HID * HID);
        }
    }

    const dim3 grid(BATCH / 64, HID / 64);
    const dim3 block(256);

    for (int l = 0; l < LAYERS; ++l) {
        const __bf16* layer_in = (l == 0) ? x_bf : (l == 1 ? seqA : seqB);

        {
            const int n = BATCH * HID;
            init_h_kernel<<<(n + 255) / 256, 256, 0, stream>>>(
                h0 + (size_t)l * BATCH * HID, h_f[0], h_b[0], n);
        }

        int ping = 0;
        for (int t = 0; t < T_STEPS; ++t) {
            const int pong = ping ^ 1;
            const int Din = (l == 0) ? IN_DIM : HID;
            const size_t tOffIn  = (size_t)t * BATCH * Din;
            const size_t tOffOut = (size_t)t * BATCH * HID;
            __bf16* seq_bf = (l == 0) ? (seqA + tOffOut)
                           : (l == 1) ? (seqB + tOffOut) : nullptr;
            float*  seq_f  = (l == 2) ? (out_seq + tOffOut) : nullptr;
            float*  hid    = (t == T_STEPS - 1) ? (out_hid + (size_t)l * BATCH * HID) : nullptr;

            if (l == 0) {
                gru_step_kernel<IN_DIM><<<grid, block, 0, stream>>>(
                    layer_in + tOffIn, wih_b[l], whh_b[l], bih[l], bhh[l],
                    h_f[ping], h_b[ping], h_f[pong], h_b[pong],
                    seq_bf, seq_f, hid);
            } else {
                gru_step_kernel<HID><<<grid, block, 0, stream>>>(
                    layer_in + tOffIn, wih_b[l], whh_b[l], bih[l], bhh[l],
                    h_f[ping], h_b[ping], h_f[pong], h_b[pong],
                    seq_bf, seq_f, hid);
            }
            ping = pong;
        }
    }
}
